// GNNStack_39479339384941
// MI455X (gfx1250) — matrix operand order repaired and checked by reading, not compiled, not run on hardware
//
#include <hip/hip_runtime.h>
#include <stdint.h>

#define N_NODES 50000
#define N_EDGES 800000
#define D 128
#define G 64
#define C 32
#define NLAYERS 3
#define LN_EPS 1e-5f

typedef __attribute__((ext_vector_type(16))) __bf16 v16bf;
typedef __attribute__((ext_vector_type(8)))  float  v8f;
typedef __attribute__((ext_vector_type(8)))  unsigned int v8u;

__device__ __forceinline__ uint32_t f2bf(float f) {
    uint32_t u = __float_as_uint(f);
    uint32_t r = u + 0x7FFFu + ((u >> 16) & 1u);   // round-to-nearest-even
    return r >> 16;
}
__device__ __forceinline__ uint32_t pack_bf2(float lo, float hi) {
    return f2bf(lo) | (f2bf(hi) << 16);
}

// ---------------------------------------------------------------- convert x -> bf16 packed
__global__ __launch_bounds__(256) void k_convert(const float* __restrict__ x,
                                                 uint32_t* __restrict__ xb, int ndw) {
    int i = blockIdx.x * 256 + threadIdx.x;
    if (i < ndw) xb[i] = pack_bf2(x[2 * i], x[2 * i + 1]);
}

// ---------------------------------------------------------------- repack weights into WMMA B layout
// Wp index = ((mat*4 + kstep)*8 + ntile)*32 + lane, 8 dwords each.
// B k order used here (contiguous, same as A; 16-bit, 32x16): lane<16 -> K = kstep*32 + 0..15 (pairs in V0..V7),
//                           lane>=16 -> K = kstep*32 + 16..31 ; n = ntile*16 + lane%16
__global__ __launch_bounds__(256) void k_repack(const float* __restrict__ Wself,
                                                const float* __restrict__ Wmsg,
                                                uint32_t* __restrict__ Wp) {
    int t = blockIdx.x * 256 + threadIdx.x;          // 0..6143
    if (t >= 6 * 4 * 8 * 32) return;
    int lane = t & 31;
    int ks   = (t >> 8) & 3;
    int mat  = t >> 10;                              // 0..5 : layer*2 + {self,msg}
    int layer = mat >> 1;
    const float* W = (mat & 1) ? (Wmsg + layer * D * D) : (Wself + layer * D * D);
    int n  = ((t >> 5) & 7) * 16 + (lane & 15);
    int kb = ks * 32 + ((lane < 16) ? 0 : 16);
    uint32_t* o = Wp + (size_t)t * 8;
#pragma unroll
    for (int v = 0; v < 8; ++v) {
        int k = kb + 2 * v;
        o[v] = pack_bf2(W[k * D + n], W[(k + 1) * D + n]);
    }
}

// ---------------------------------------------------------------- dual GEMM: self & msg linears
// grid = N/16 blocks, 256 thr (8 waves). Wave w owns N-tile w (16 cols); block owns 16 rows.
__global__ __launch_bounds__(256) void k_gemm2(const uint32_t* __restrict__ xb,
                                               const uint32_t* __restrict__ Wp,
                                               const float* __restrict__ bs,
                                               const float* __restrict__ bm,
                                               float* __restrict__ outs,
                                               float* __restrict__ outm,
                                               int layer) {
    const int lane = threadIdx.x & 31;
    const int wv   = threadIdx.x >> 5;                       // ntile 0..7
    const int mrow = blockIdx.x * 16 + (lane & 15);
    const int kbA  = (lane < 16) ? 0 : 16;                   // A half-select (elements)
    const uint32_t* arow = xb + (size_t)mrow * (D / 2);
    v8f cs = {}; v8f cm = {};
#pragma unroll
    for (int ks = 0; ks < 4; ++ks) {
        int ad = (ks * 32 + kbA) >> 1;                       // dword offset in row
        uint4 a0 = *(const uint4*)(arow + ad);
        uint4 a1 = *(const uint4*)(arow + ad + 4);
        v8u au; au[0]=a0.x; au[1]=a0.y; au[2]=a0.z; au[3]=a0.w;
                au[4]=a1.x; au[5]=a1.y; au[6]=a1.z; au[7]=a1.w;
        v16bf A = __builtin_bit_cast(v16bf, au);

        const uint32_t* ps = Wp + ((((size_t)(layer*2+0)*4 + ks)*8 + wv)*32 + lane)*8;
        const uint32_t* pm = Wp + ((((size_t)(layer*2+1)*4 + ks)*8 + wv)*32 + lane)*8;
        uint4 s0 = *(const uint4*)(ps);
        uint4 s1 = *(const uint4*)(ps + 4);
        uint4 m0 = *(const uint4*)(pm);
        uint4 m1 = *(const uint4*)(pm + 4);
        v8u bu; bu[0]=s0.x; bu[1]=s0.y; bu[2]=s0.z; bu[3]=s0.w;
                bu[4]=s1.x; bu[5]=s1.y; bu[6]=s1.z; bu[7]=s1.w;
        v16bf Bs = __builtin_bit_cast(v16bf, bu);
        v8u cu; cu[0]=m0.x; cu[1]=m0.y; cu[2]=m0.z; cu[3]=m0.w;
                cu[4]=m1.x; cu[5]=m1.y; cu[6]=m1.z; cu[7]=m1.w;
        v16bf Bm = __builtin_bit_cast(v16bf, cu);

        cs = __builtin_amdgcn_wmma_f32_16x16x32_bf16(false, A, false, Bs, (short)0, cs, false, false);
        cm = __builtin_amdgcn_wmma_f32_16x16x32_bf16(false, A, false, Bm, (short)0, cm, false, false);
    }
    int n = wv * 16 + (lane & 15);
    float bsv = bs[layer * D + n];
    float bmv = bm[layer * D + n];
    int rbase = blockIdx.x * 16 + ((lane >= 16) ? 8 : 0);
#pragma unroll
    for (int r = 0; r < 8; ++r) {
        outs[(size_t)(rbase + r) * D + n] = cs[r] + bsv;
        outm[(size_t)(rbase + r) * D + n] = cm[r] + bmv;
    }
}

// ---------------------------------------------------------------- edge scatter-add (agg += msg[src] @ dst)
__global__ __launch_bounds__(256) void k_scatter(const float4* __restrict__ msg4,
                                                 const int* __restrict__ src,
                                                 const int* __restrict__ dst,
                                                 float* __restrict__ agg) {
    int idx = blockIdx.x * 256 + threadIdx.x;
    int e = idx >> 5;
    int c = idx & 31;
    if (e >= N_EDGES) return;
    int s = src[e], d = dst[e];
    float4 v = msg4[(size_t)s * 32 + c];
    float* a = agg + (size_t)d * D + c * 4;
    atomicAdd(a + 0, v.x); atomicAdd(a + 1, v.y);
    atomicAdd(a + 2, v.z); atomicAdd(a + 3, v.w);
}

// ---------------------------------------------------------------- combine: relu(self+agg), optional LN, emit fp32/bf16
__global__ __launch_bounds__(256) void k_combine(const float* __restrict__ so,
                                                 const float* __restrict__ agg,
                                                 const float* __restrict__ lng,
                                                 const float* __restrict__ lnb,
                                                 float* __restrict__ hout,
                                                 uint32_t* __restrict__ xb,
                                                 int do_ln) {
    int lane = threadIdx.x & 31;
    int node = blockIdx.x * 8 + (threadIdx.x >> 5);
    size_t base = (size_t)node * D + lane * 4;
    float4 a = *(const float4*)(so + base);
    float4 g = *(const float4*)(agg + base);
    float v0 = fmaxf(a.x + g.x, 0.f), v1 = fmaxf(a.y + g.y, 0.f);
    float v2 = fmaxf(a.z + g.z, 0.f), v3 = fmaxf(a.w + g.w, 0.f);
    if (do_ln) {
        float s = v0 + v1 + v2 + v3;
#pragma unroll
        for (int m = 16; m; m >>= 1) s += __shfl_xor(s, m, 32);
        float mu = s * (1.0f / 128.0f);
        float d0 = v0 - mu, d1 = v1 - mu, d2 = v2 - mu, d3 = v3 - mu;
        float q = d0 * d0 + d1 * d1 + d2 * d2 + d3 * d3;
#pragma unroll
        for (int m = 16; m; m >>= 1) q += __shfl_xor(q, m, 32);
        float rs = rsqrtf(q * (1.0f / 128.0f) + LN_EPS);
        int f = lane * 4;
        v0 = lng[f + 0] * d0 * rs + lnb[f + 0];
        v1 = lng[f + 1] * d1 * rs + lnb[f + 1];
        v2 = lng[f + 2] * d2 * rs + lnb[f + 2];
        v3 = lng[f + 3] * d3 * rs + lnb[f + 3];
    }
    if (hout) {
        float4 o; o.x = v0; o.y = v1; o.z = v2; o.w = v3;
        *(float4*)(hout + base) = o;
    }
    if (xb) {
        xb[(size_t)node * 64 + lane * 2]     = pack_bf2(v0, v1);
        xb[(size_t)node * 64 + lane * 2 + 1] = pack_bf2(v2, v3);
    }
}

// ---------------------------------------------------------------- mean-pool accumulate
__global__ __launch_bounds__(256) void k_pool(const float* __restrict__ h,
                                              const int* __restrict__ batch,
                                              float* __restrict__ sums,
                                              float* __restrict__ cnt) {
    int lane = threadIdx.x & 31;
    int node = blockIdx.x * 8 + (threadIdx.x >> 5);
    int g = batch[node];
    float4 v = *(const float4*)(h + (size_t)node * D + lane * 4);
    float* s = sums + (size_t)g * D + lane * 4;
    atomicAdd(s + 0, v.x); atomicAdd(s + 1, v.y);
    atomicAdd(s + 2, v.z); atomicAdd(s + 3, v.w);
    if (lane == 0) atomicAdd(cnt + g, 1.0f);
}

// ---------------------------------------------------------------- head
__global__ __launch_bounds__(256) void k_pooled(const float* __restrict__ sums,
                                                const float* __restrict__ cnt,
                                                float* __restrict__ pooled) {
    int i = blockIdx.x * 256 + threadIdx.x;          // G*D = 8192
    int g = i >> 7;
    pooled[i] = sums[i] / fmaxf(cnt[g], 1.0f);
}

__global__ __launch_bounds__(256) void k_fc1(const float* __restrict__ pooled,
                                             const float* __restrict__ W1,
                                             const float* __restrict__ b1,
                                             float* __restrict__ h) {
    int i = blockIdx.x * 256 + threadIdx.x;          // 8192
    int g = i >> 7, n = i & 127;
    float acc = b1[n];
    const float* p = pooled + g * D;
#pragma unroll 8
    for (int k = 0; k < D; ++k) acc = fmaf(p[k], W1[k * D + n], acc);
    h[i] = acc;
}

__global__ __launch_bounds__(256) void k_fc2(const float* __restrict__ h,
                                             const float* __restrict__ W2,
                                             const float* __restrict__ b2,
                                             float* __restrict__ logits) {
    int i = blockIdx.x * 256 + threadIdx.x;          // G*C = 2048
    int g = i >> 5, c = i & 31;
    float acc = b2[c];
    const float* p = h + g * D;
#pragma unroll 8
    for (int k = 0; k < D; ++k) acc = fmaf(p[k], W2[k * C + c], acc);
    logits[i] = acc;
}

__global__ __launch_bounds__(256) void k_lsm(const float* __restrict__ logits,
                                             float* __restrict__ out) {
    int lane = threadIdx.x & 31;                     // one lane per class (C==32)
    int g = blockIdx.x * 8 + (threadIdx.x >> 5);
    float x = logits[g * C + lane];
    float mx = x;
#pragma unroll
    for (int m = 16; m; m >>= 1) mx = fmaxf(mx, __shfl_xor(mx, m, 32));
    float ex = expf(x - mx);
    float s = ex;
#pragma unroll
    for (int m = 16; m; m >>= 1) s += __shfl_xor(s, m, 32);
    out[g * C + lane] = (x - mx) - logf(s);
}

// ----------------------------------------------------------------
extern "C" void kernel_launch(void* const* d_in, const int* in_sizes, int n_in,
                              void* d_out, int out_size, void* d_ws, size_t ws_size,
                              hipStream_t stream) {
    const float* x      = (const float*)d_in[0];
    const float* W_self = (const float*)d_in[1];
    const float* b_self = (const float*)d_in[2];
    const float* W_msg  = (const float*)d_in[3];
    const float* b_msg  = (const float*)d_in[4];
    const float* ln_g   = (const float*)d_in[5];
    const float* ln_b   = (const float*)d_in[6];
    const float* W1     = (const float*)d_in[7];
    const float* b1     = (const float*)d_in[8];
    const float* W2     = (const float*)d_in[9];
    const float* b2     = (const float*)d_in[10];
    const int*   edge   = (const int*)d_in[11];
    const int*   batch  = (const int*)d_in[12];
    const int* src = edge;
    const int* dst = edge + N_EDGES;

    char* w = (char*)d_ws;
    auto carve = [&](size_t bytes) { char* p = w; w += (bytes + 255) & ~(size_t)255; return p; };
    float*    hfin   = (float*)   carve((size_t)N_NODES * D * 4);  // final-layer fp32 activations
    float*    so     = (float*)   carve((size_t)N_NODES * D * 4);  // self-linear out
    float*    msg    = (float*)   carve((size_t)N_NODES * D * 4);  // msg-linear out
    float*    agg    = (float*)   carve((size_t)N_NODES * D * 4);  // scatter target
    uint32_t* xb     = (uint32_t*)carve((size_t)N_NODES * D * 2);  // bf16 activations (packed)
    uint32_t* Wp     = (uint32_t*)carve((size_t)6 * D * D / 2 * 4);
    float*    sums   = (float*)   carve((size_t)G * D * 4);
    float*    cnt    = (float*)   carve((size_t)G * 4);
    float*    pooled = (float*)   carve((size_t)G * D * 4);
    float*    hh     = (float*)   carve((size_t)G * D * 4);
    float*    logits = (float*)   carve((size_t)G * C * 4);

    k_repack <<<24, 256, 0, stream>>>(W_self, W_msg, Wp);
    k_convert<<<(N_NODES * D / 2 + 255) / 256, 256, 0, stream>>>(x, xb, N_NODES * D / 2);

    for (int layer = 0; layer < NLAYERS; ++layer) {
        k_gemm2<<<N_NODES / 16, 256, 0, stream>>>(xb, Wp, b_self, b_msg, so, msg, layer);
        hipMemsetAsync(agg, 0, (size_t)N_NODES * D * 4, stream);
        k_scatter<<<N_EDGES * 32 / 256, 256, 0, stream>>>((const float4*)msg, src, dst, agg);
        int last = (layer == NLAYERS - 1);
        k_combine<<<N_NODES / 8, 256, 0, stream>>>(
            so, agg, ln_g + (last ? 0 : layer) * D, ln_b + (last ? 0 : layer) * D,
            last ? hfin : nullptr, last ? nullptr : xb, last ? 0 : 1);
    }

    hipMemsetAsync(sums, 0, (size_t)G * D * 4, stream);
    hipMemsetAsync(cnt,  0, (size_t)G * 4, stream);
    k_pool  <<<N_NODES / 8, 256, 0, stream>>>(hfin, batch, sums, cnt);
    k_pooled<<<(G * D) / 256, 256, 0, stream>>>(sums, cnt, pooled);
    k_fc1   <<<(G * D) / 256, 256, 0, stream>>>(pooled, W1, b1, hh);
    k_fc2   <<<(G * C) / 256, 256, 0, stream>>>(hh, W2, b2, logits);
    k_lsm   <<<G / 8, 256, 0, stream>>>(logits, (float*)d_out);
}
